// GPT_35055523070439
// MI455X (gfx1250) — compile-verified, matrix operand order checked by reading, not run on hardware
//
#include <hip/hip_runtime.h>
#include <math.h>

// ---------------------------------------------------------------------------
// GPT-2 (L=6, H=12, C=768, B=2, T=1024, V=50257) forward on gfx1250 (CDNA5)
// All matmuls via v_wmma_f32_16x16x32_bf16 (wave32 WMMA).
// Straight-copy LDS staging uses GLOBAL_LOAD_ASYNC_TO_LDS_B128 (ASYNCcnt).
// ---------------------------------------------------------------------------

#define Cc   768
#define Tt   1024
#define Bb   2
#define Mrows (Bb*Tt)      // 2048 token rows
#define Hh   12
#define HD   64
#define Ll   6
#define Vv   50257
#define Vpad 50304         // 393 * 128

typedef unsigned short u16;
typedef __attribute__((ext_vector_type(16))) __bf16 v16bf;
typedef __attribute__((ext_vector_type(8)))  float  v8f;
typedef __attribute__((ext_vector_type(8)))  unsigned short ushort8;

union FragB { v16bf v; ushort8 h[2]; };

static __device__ inline u16 f2bf(float f) {
  unsigned u = __builtin_bit_cast(unsigned, f);
  u += 0x7fffu + ((u >> 16) & 1u);          // round-to-nearest-even
  return (u16)(u >> 16);
}

static __device__ inline float geluf(float v) {
  return 0.5f * v * (1.0f + tanhf(0.7978845608028654f * (v + 0.044715f * v * v * v)));
}

// async copy of 32 bytes (2 x b128) global -> LDS; tracked with ASYNCcnt
static __device__ inline void async_copy32(void* lds_dst, const void* gsrc) {
  unsigned ldsoff = (unsigned)(uintptr_t)lds_dst;       // low 32 bits == LDS offset
  unsigned long long ga = (unsigned long long)(uintptr_t)gsrc;
  asm volatile(
      "global_load_async_to_lds_b128 %0, %1, off\n\t"
      "global_load_async_to_lds_b128 %0, %1, off offset:16"
      :: "v"(ldsoff), "v"(ga) : "memory");
}

// async copy of 64 bytes (4 x b128) global -> LDS
static __device__ inline void async_copy64(void* lds_dst, const void* gsrc) {
  unsigned ldsoff = (unsigned)(uintptr_t)lds_dst;
  unsigned long long ga = (unsigned long long)(uintptr_t)gsrc;
  asm volatile(
      "global_load_async_to_lds_b128 %0, %1, off\n\t"
      "global_load_async_to_lds_b128 %0, %1, off offset:16\n\t"
      "global_load_async_to_lds_b128 %0, %1, off offset:32\n\t"
      "global_load_async_to_lds_b128 %0, %1, off offset:48"
      :: "v"(ldsoff), "v"(ga) : "memory");
}

static __device__ inline void wait_async0() {
  asm volatile("s_wait_asynccnt 0x0" ::: "memory");
}

// ---------------------------------------------------------------------------
// Embedding: x[b,t,:] = wte[idx[b,t],:] + wpe[t,:]   (fp32 residual stream)
// ---------------------------------------------------------------------------
__global__ __launch_bounds__(256) void embed_kernel(
    const int* __restrict__ idx, const float* __restrict__ wte,
    const float* __restrict__ wpe, float* __restrict__ x) {
  int i = blockIdx.x * 256 + threadIdx.x;      // < 2048*768
  int tok = i / Cc, c = i - tok * Cc;
  int t = tok & (Tt - 1);
  x[i] = wte[(size_t)idx[tok] * Cc + c] + wpe[(size_t)t * Cc + c];
}

// ---------------------------------------------------------------------------
// LayerNorm over C=768, output bf16 activations. One row per block.
// ---------------------------------------------------------------------------
__global__ __launch_bounds__(256) void ln_kernel(
    const float* __restrict__ x, const float* __restrict__ s,
    const float* __restrict__ b, u16* __restrict__ out) {
  __shared__ float red[256];
  const int row = blockIdx.x, tid = threadIdx.x;
  const float* xr = x + (size_t)row * Cc;
  float v0 = xr[tid], v1 = xr[tid + 256], v2 = xr[tid + 512];
  red[tid] = v0 + v1 + v2;
  __syncthreads();
  for (int o = 128; o > 0; o >>= 1) { if (tid < o) red[tid] += red[tid + o]; __syncthreads(); }
  float mu = red[0] * (1.0f / Cc);
  __syncthreads();
  float d0 = v0 - mu, d1 = v1 - mu, d2 = v2 - mu;
  red[tid] = d0 * d0 + d1 * d1 + d2 * d2;
  __syncthreads();
  for (int o = 128; o > 0; o >>= 1) { if (tid < o) red[tid] += red[tid + o]; __syncthreads(); }
  float rstd = rsqrtf(red[0] * (1.0f / Cc) + 1e-5f);
  u16* orow = out + (size_t)row * Cc;
  orow[tid]       = f2bf(d0 * rstd * s[tid]       + b[tid]);
  orow[tid + 256] = f2bf(d1 * rstd * s[tid + 256] + b[tid + 256]);
  orow[tid + 512] = f2bf(d2 * rstd * s[tid + 512] + b[tid + 512]);
}

// ---------------------------------------------------------------------------
// Tiled WMMA GEMM: out[M,N] = A[M,KDIM](bf16) @ W[KDIM,Nw](f32, converted)
//   block tile 128x128, k-step 32, 8 waves: wave = (m 0..1) x (n 0..3),
//   each wave does 4x2 16x16 tiles -> 8 wmma per k-step.
// LDS: As[128][40] bf16 (row-major, async-staged), Bs[128][40] bf16 ([n][k]).
// ---------------------------------------------------------------------------
template <int KDIM, bool BIAS, bool GELU, bool RES, bool OUTF32, bool OUTBF16>
__global__ __launch_bounds__(256) void gemm_kernel(
    const u16* __restrict__ A, const float* __restrict__ W,
    const float* __restrict__ bias, const float* __restrict__ res,
    float* __restrict__ outf, u16* __restrict__ outb, int Nw) {
  __shared__ u16 As[128 * 40];
  __shared__ u16 Bs[128 * 40];
  const int nblk = blockIdx.x * 128, mblk = blockIdx.y * 128;
  const int tid = threadIdx.x, wave = tid >> 5, lane = tid & 31;
  const int lr = lane & 15, kh = lane >> 4;
  const int m0 = (wave >> 2) * 64, n0 = (wave & 3) * 32;

  v8f acc[4][2] = {};

  for (int kb = 0; kb < KDIM; kb += 32) {
    __syncthreads();
    // ---- stage A: 128 rows x 32 k, async straight copy (bf16) ----
    {
      int row = tid >> 1, half = tid & 1;
      const u16* src = A + (size_t)(mblk + row) * KDIM + kb + half * 16;
      async_copy32(&As[row * 40 + half * 16], src);
    }
    // ---- stage B transposed: Bs[n][k] = bf16(W[kb+k][nblk+n]) ----
    {
      int k = tid >> 3, nc = (tid & 7) * 16;
      const float* wrow = W + (size_t)(kb + k) * Nw + nblk + nc;
      if (kb + 32 < KDIM) __builtin_prefetch(wrow + (size_t)32 * Nw, 0, 0);
#pragma unroll
      for (int i = 0; i < 16; ++i) {
        int gn = nblk + nc + i;
        float w = (gn < Nw) ? wrow[i] : 0.0f;
        Bs[(nc + i) * 40 + k] = f2bf(w);
      }
    }
    wait_async0();
    __syncthreads();
    // ---- fragments + WMMA ----
    FragB afr[4], bfr[2];
#pragma unroll
    for (int i = 0; i < 4; ++i) {
      const u16* base = &As[(m0 + i * 16 + lr) * 40];
      afr[i].h[0] = *(const ushort8*)(base + kh * 8);
      afr[i].h[1] = *(const ushort8*)(base + 16 + kh * 8);
    }
#pragma unroll
    for (int j = 0; j < 2; ++j) {
      const u16* base = &Bs[(n0 + j * 16 + lr) * 40];
      bfr[j].h[0] = *(const ushort8*)(base + kh * 8);
      bfr[j].h[1] = *(const ushort8*)(base + 16 + kh * 8);
    }
#pragma unroll
    for (int i = 0; i < 4; ++i)
#pragma unroll
      for (int j = 0; j < 2; ++j)
        acc[i][j] = __builtin_amdgcn_wmma_f32_16x16x32_bf16(
            false, afr[i].v, false, bfr[j].v, (short)0, acc[i][j], false, false);
  }

  // ---- epilogue: C/D layout -> lane holds col (lane&15), rows r + 8*(lane>>4)
#pragma unroll
  for (int i = 0; i < 4; ++i) {
#pragma unroll
    for (int j = 0; j < 2; ++j) {
      int col = nblk + n0 + j * 16 + lr;
      float bv = 0.0f;
      if (BIAS && col < Nw) bv = bias[col];
#pragma unroll
      for (int r = 0; r < 8; ++r) {
        int row = mblk + m0 + i * 16 + kh * 8 + r;
        float v = acc[i][j][r] + bv;
        if (GELU) v = geluf(v);
        if (RES && col < Nw) v += res[(size_t)row * Nw + col];
        if (OUTF32 && col < Nw) outf[(size_t)row * Nw + col] = v;
        if (OUTBF16 && col < Nw) outb[(size_t)row * Nw + col] = f2bf(v);
      }
    }
  }
}

// ---------------------------------------------------------------------------
// Flash attention: one block per (qblock128, head, batch). 8 waves, each wave
// owns 16 q rows. K staged row-major via async LDS loads (== B-fragment layout
// for Q@K^T), V staged transposed, P re-laid out via per-wave LDS for P@V.
// ---------------------------------------------------------------------------
__global__ __launch_bounds__(256) void attn_kernel(
    const u16* __restrict__ qkv, u16* __restrict__ y) {
  __shared__ u16 Kb[128 * 72];       // [kv][hd]
  __shared__ u16 Vt[64 * 136];       // [hd][kv]
  __shared__ u16 Pb[8 * 16 * 72];    // per-wave P halves [16][64+pad]
  const int qb = blockIdx.x, hh = blockIdx.y, bb = blockIdx.z;
  const int tid = threadIdx.x, wave = tid >> 5, lane = tid & 31;
  const int lr = lane & 15, kh = lane >> 4;
  const int qrow0 = qb * 128 + wave * 16;
  u16* Pw = Pb + wave * 16 * 72;

  // Q fragments (2 k-chunks of 32 over hd=64), straight from global bf16 qkv
  FragB qf[2];
  {
    const u16* qp = qkv + (size_t)(bb * Tt + qrow0 + lr) * (3 * Cc) + hh * HD;
#pragma unroll
    for (int kc = 0; kc < 2; ++kc) {
      qf[kc].h[0] = *(const ushort8*)(qp + kc * 32 + kh * 8);
      qf[kc].h[1] = *(const ushort8*)(qp + kc * 32 + 16 + kh * 8);
    }
  }

  v8f O[4] = {};
  float mrow[8], lrow[8];
#pragma unroll
  for (int r = 0; r < 8; ++r) { mrow[r] = -1e30f; lrow[r] = 0.0f; }

  for (int jb = 0; jb <= qb; ++jb) {
    __syncthreads();
    // ---- stage K (row-major, async) and V (transposed, manual) ----
    {
      int r = tid >> 1, half = tid & 1;
      const u16* kp = qkv + (size_t)(bb * Tt + jb * 128 + r) * (3 * Cc) + Cc + hh * HD + half * 32;
      async_copy64(&Kb[r * 72 + half * 32], kp);
      const u16* vp = qkv + (size_t)(bb * Tt + jb * 128 + r) * (3 * Cc) + 2 * Cc + hh * HD + half * 32;
#pragma unroll
      for (int i = 0; i < 32; ++i) Vt[(half * 32 + i) * 136 + r] = vp[i];
    }
    wait_async0();
    __syncthreads();

    // ---- S = Q @ K^T  (8 tiles of 16 kv cols) ----
    v8f s[8];
#pragma unroll
    for (int nt = 0; nt < 8; ++nt) {
      const u16* kbp = Kb + (nt * 16 + lr) * 72;
      FragB k0, k1;
      k0.h[0] = *(const ushort8*)(kbp + kh * 8);
      k0.h[1] = *(const ushort8*)(kbp + 16 + kh * 8);
      k1.h[0] = *(const ushort8*)(kbp + 32 + kh * 8);
      k1.h[1] = *(const ushort8*)(kbp + 48 + kh * 8);
      v8f z = {};
      z = __builtin_amdgcn_wmma_f32_16x16x32_bf16(false, qf[0].v, false, k0.v, (short)0, z, false, false);
      z = __builtin_amdgcn_wmma_f32_16x16x32_bf16(false, qf[1].v, false, k1.v, (short)0, z, false, false);
      s[nt] = z;
    }

    // ---- scale + causal mask + online softmax stats ----
    float mb[8];
#pragma unroll
    for (int r = 0; r < 8; ++r) mb[r] = -1e30f;
#pragma unroll
    for (int nt = 0; nt < 8; ++nt) {
#pragma unroll
      for (int r = 0; r < 8; ++r) {
        float v = s[nt][r] * 0.125f;                 // 1/sqrt(64)
        int qrow = qrow0 + kh * 8 + r;
        int kcol = jb * 128 + nt * 16 + lr;
        if (kcol > qrow) v = -1e30f;
        s[nt][r] = v;
        mb[r] = fmaxf(mb[r], v);
      }
    }
#pragma unroll
    for (int msk = 1; msk < 16; msk <<= 1)
#pragma unroll
      for (int r = 0; r < 8; ++r) mb[r] = fmaxf(mb[r], __shfl_xor(mb[r], msk, 32));

    float al[8], mn[8], ls[8];
#pragma unroll
    for (int r = 0; r < 8; ++r) {
      mn[r] = fmaxf(mrow[r], mb[r]);
      al[r] = __expf(mrow[r] - mn[r]);
      mrow[r] = mn[r];
      lrow[r] *= al[r];
      ls[r] = 0.0f;
    }
#pragma unroll
    for (int nt = 0; nt < 8; ++nt)
#pragma unroll
      for (int r = 0; r < 8; ++r) {
        float p = __expf(s[nt][r] - mn[r]);
        s[nt][r] = p;
        ls[r] += p;
      }
#pragma unroll
    for (int msk = 1; msk < 16; msk <<= 1)
#pragma unroll
      for (int r = 0; r < 8; ++r) ls[r] += __shfl_xor(ls[r], msk, 32);
#pragma unroll
    for (int r = 0; r < 8; ++r) lrow[r] += ls[r];
#pragma unroll
    for (int n = 0; n < 4; ++n)
#pragma unroll
      for (int r = 0; r < 8; ++r) O[n][r] *= al[r];

    // ---- O += P @ V, in two 64-kv halves through per-wave LDS ----
#pragma unroll
    for (int half = 0; half < 2; ++half) {
#pragma unroll
      for (int nt2 = 0; nt2 < 4; ++nt2) {
        int nt = half * 4 + nt2;
#pragma unroll
        for (int r = 0; r < 8; ++r)
          Pw[(kh * 8 + r) * 72 + nt2 * 16 + lr] = f2bf(s[nt][r]);
      }
      asm volatile("s_wait_dscnt 0" ::: "memory");
#pragma unroll
      for (int kc = 0; kc < 2; ++kc) {
        FragB pa;
        const u16* pp = Pw + lr * 72 + kc * 32;
        pa.h[0] = *(const ushort8*)(pp + kh * 8);
        pa.h[1] = *(const ushort8*)(pp + 16 + kh * 8);
        int kk = half * 2 + kc;
#pragma unroll
        for (int n = 0; n < 4; ++n) {
          FragB vb;
          const u16* vtp = Vt + (n * 16 + lr) * 136 + kk * 32;
          vb.h[0] = *(const ushort8*)(vtp + kh * 8);
          vb.h[1] = *(const ushort8*)(vtp + 16 + kh * 8);
          O[n] = __builtin_amdgcn_wmma_f32_16x16x32_bf16(
              false, pa.v, false, vb.v, (short)0, O[n], false, false);
        }
      }
    }
  }

  // ---- normalize + write y (bf16, [tok][h*64+hd]) ----
#pragma unroll
  for (int n = 0; n < 4; ++n) {
#pragma unroll
    for (int r = 0; r < 8; ++r) {
      int qrow = qrow0 + kh * 8 + r;
      float v = O[n][r] / lrow[r];
      y[(size_t)(bb * Tt + qrow) * Cc + hh * HD + n * 16 + lr] = f2bf(v);
    }
  }
}

// ---------------------------------------------------------------------------
// Cross-entropy: per-row logsumexp over V=50257, then masked mean.
// ---------------------------------------------------------------------------
__global__ __launch_bounds__(256) void loss_rows_kernel(
    const float* __restrict__ logits, const int* __restrict__ tg,
    float* __restrict__ nll) {
  __shared__ float red[256];
  const int row = blockIdx.x, tid = threadIdx.x;
  const float* lg = logits + (size_t)row * Vv;
  float m = -3.4e38f;
  for (int c = tid; c < Vv; c += 256) m = fmaxf(m, lg[c]);
  red[tid] = m;
  __syncthreads();
  for (int o = 128; o > 0; o >>= 1) { if (tid < o) red[tid] = fmaxf(red[tid], red[tid + o]); __syncthreads(); }
  m = red[0];
  __syncthreads();
  float sum = 0.0f;
  for (int c = tid; c < Vv; c += 256) sum += __expf(lg[c] - m);
  red[tid] = sum;
  __syncthreads();
  for (int o = 128; o > 0; o >>= 1) { if (tid < o) red[tid] += red[tid + o]; __syncthreads(); }
  if (tid == 0) {
    int t = tg[row];
    int tc = (t < 0) ? 0 : t;
    float lp = lg[tc] - m - __logf(red[0]);
    nll[row] = (t != -1) ? -lp : 0.0f;
  }
}

__global__ __launch_bounds__(256) void loss_final_kernel(
    const float* __restrict__ nll, const int* __restrict__ tg,
    float* __restrict__ loss) {
  __shared__ float rs[256];
  __shared__ float rc[256];
  const int tid = threadIdx.x;
  float s = 0.0f, c = 0.0f;
  for (int i = tid; i < Mrows; i += 256) {
    s += nll[i];
    c += (tg[i] != -1) ? 1.0f : 0.0f;
  }
  rs[tid] = s; rc[tid] = c;
  __syncthreads();
  for (int o = 128; o > 0; o >>= 1) {
    if (tid < o) { rs[tid] += rs[tid + o]; rc[tid] += rc[tid + o]; }
    __syncthreads();
  }
  if (tid == 0) loss[0] = rs[0] / rc[0];
}

// ---------------------------------------------------------------------------
extern "C" void kernel_launch(void* const* d_in, const int* in_sizes, int n_in,
                              void* d_out, int out_size, void* d_ws, size_t ws_size,
                              hipStream_t stream) {
  const int*   idx         = (const int*)d_in[0];
  const int*   targets     = (const int*)d_in[1];
  const float* wte         = (const float*)d_in[2];
  const float* wpe         = (const float*)d_in[3];
  const float* c_attn_w    = (const float*)d_in[4];
  const float* c_attn_b    = (const float*)d_in[5];
  const float* attn_proj_w = (const float*)d_in[6];
  const float* attn_proj_b = (const float*)d_in[7];
  const float* ln1_s       = (const float*)d_in[8];
  const float* ln1_b       = (const float*)d_in[9];
  const float* ln2_s       = (const float*)d_in[10];
  const float* ln2_b       = (const float*)d_in[11];
  const float* fc_w        = (const float*)d_in[12];
  const float* fc_b        = (const float*)d_in[13];
  const float* mlp_proj_w  = (const float*)d_in[14];
  const float* mlp_proj_b  = (const float*)d_in[15];
  const float* lnf_s       = (const float*)d_in[16];
  const float* lnf_b       = (const float*)d_in[17];
  const float* lm_head_w   = (const float*)d_in[18];

  // workspace carve-up (~35 MB)
  char* w = (char*)d_ws;
  float* x    = (float*)w;  w += (size_t)Mrows * Cc * 4;        // fp32 residual
  u16*   h    = (u16*)w;    w += (size_t)Mrows * Cc * 2;        // bf16 LN output
  u16*   qkvb = (u16*)w;    w += (size_t)Mrows * 3 * Cc * 2;    // bf16 qkv
  u16*   yb   = (u16*)w;    w += (size_t)Mrows * Cc * 2;        // bf16 attn out
  u16*   ab   = (u16*)w;    w += (size_t)Mrows * 4 * Cc * 2;    // bf16 fc out
  float* nll  = (float*)w;  w += (size_t)Mrows * 4;

  float* logits = (float*)d_out;
  float* loss   = logits + (size_t)Mrows * Vv;

  // 1) embedding
  embed_kernel<<<(Mrows * Cc) / 256, 256, 0, stream>>>(idx, wte, wpe, x);

  for (int l = 0; l < Ll; ++l) {
    // 2) ln1 -> h (bf16)
    ln_kernel<<<Mrows, 256, 0, stream>>>(x, ln1_s + (size_t)l * Cc, ln1_b + (size_t)l * Cc, h);
    // 3) qkv = h @ Wqkv + b   (bf16 out)
    gemm_kernel<Cc, true, false, false, false, true>
        <<<dim3(3 * Cc / 128, Mrows / 128), 256, 0, stream>>>(
            h, c_attn_w + (size_t)l * Cc * 3 * Cc, c_attn_b + (size_t)l * 3 * Cc,
            nullptr, nullptr, qkvb, 3 * Cc);
    // 4) flash attention -> yb (bf16)
    attn_kernel<<<dim3(Tt / 128, Hh, Bb), 256, 0, stream>>>(qkvb, yb);
    // 5) x += yb @ Wproj + b  (fp32, in-place residual)
    gemm_kernel<Cc, true, false, true, true, false>
        <<<dim3(Cc / 128, Mrows / 128), 256, 0, stream>>>(
            yb, attn_proj_w + (size_t)l * Cc * Cc, attn_proj_b + (size_t)l * Cc,
            x, x, nullptr, Cc);
    // 6) ln2 -> h
    ln_kernel<<<Mrows, 256, 0, stream>>>(x, ln2_s + (size_t)l * Cc, ln2_b + (size_t)l * Cc, h);
    // 7) ab = gelu(h @ Wfc + b)  (bf16 out)
    gemm_kernel<Cc, true, true, false, false, true>
        <<<dim3(4 * Cc / 128, Mrows / 128), 256, 0, stream>>>(
            h, fc_w + (size_t)l * Cc * 4 * Cc, fc_b + (size_t)l * 4 * Cc,
            nullptr, nullptr, ab, 4 * Cc);
    // 8) x += ab @ Wmlp + b  (fp32, in-place residual)
    gemm_kernel<4 * Cc, true, false, true, true, false>
        <<<dim3(Cc / 128, Mrows / 128), 256, 0, stream>>>(
            ab, mlp_proj_w + (size_t)l * 4 * Cc * Cc, mlp_proj_b + (size_t)l * Cc,
            x, x, nullptr, Cc);
  }

  // 9) final LN -> h
  ln_kernel<<<Mrows, 256, 0, stream>>>(x, lnf_s, lnf_b, h);
  // 10) logits = h @ lm_head_w   (N padded 50257->50304, bounds-checked stores)
  gemm_kernel<Cc, false, false, false, true, false>
      <<<dim3(Vpad / 128, Mrows / 128), 256, 0, stream>>>(
          h, lm_head_w, nullptr, nullptr, logits, nullptr, Vv);
  // 11) loss
  loss_rows_kernel<<<Mrows, 256, 0, stream>>>(logits, targets, nll);
  loss_final_kernel<<<1, 256, 0, stream>>>(nll, targets, loss);
}
